// SimpleLSTMModel_80745385164834
// MI455X (gfx1250) — compile-verified, matrix operand order checked by reading, not run on hardware
//
#include <hip/hip_runtime.h>

#define BB 8
#define NN 2048
#define DD 64
#define HH 64
#define OO 64
#define NEGV (-9e15f)

typedef __attribute__((ext_vector_type(16))) __bf16 v16bf;
typedef __attribute__((ext_vector_type(8)))  float  v8f;

union AF { unsigned u[8]; uint4 q[2]; v16bf v; };

// workspace layout (in dwords/floats)
#define OFF_WH    0u
#define OFF_ASRC  1048576u
#define OFF_ADST  1064960u
#define OFF_XH    1081344u
#define OFF_WHB   2129920u
#define OFF_GW    2654208u
#define OFF_GU    2662400u
#define OFF_WOUT  2670592u

__device__ __forceinline__ unsigned pkbf(float lo, float hi) {
  unsigned a = __float_as_uint(lo); a = (a + 0x7FFFu + ((a >> 16) & 1u)) >> 16;
  unsigned b = __float_as_uint(hi); b = (b + 0x7FFFu + ((b >> 16) & 1u)) >> 16;
  return a | (b << 16);
}

__device__ __forceinline__ v8f v8zero() {
  v8f z;
#pragma unroll
  for (int i = 0; i < 8; ++i) z[i] = 0.f;
  return z;
}

__device__ __forceinline__ v8f wmma_bf(const AF& a, const AF& b, v8f c) {
  return __builtin_amdgcn_wmma_f32_16x16x32_bf16(
      false, a.v, false, b.v, (short)0, c, false, false);
}

// ---------------- Kernel A: Wh = x@W, a_src = Wh@a[:H], a_dst = Wh@a[H:] ----
__global__ void k_wh(const float* __restrict__ x, const float* __restrict__ W,
                     const float* __restrict__ av, float* __restrict__ wh,
                     float* __restrict__ asrc, float* __restrict__ adst) {
  __shared__ float sx[4 * 64];
  __shared__ float2 red[256];
  int t = threadIdx.x;
  int lr = t >> 6, col = t & 63;
  long R = (long)blockIdx.x * 4 + lr;          // flattened row over B*N
  sx[t] = x[R * DD + col];
  __syncthreads();
  float acc = 0.f;
#pragma unroll
  for (int k = 0; k < DD; ++k) acc += sx[lr * 64 + k] * W[k * HH + col];
  wh[R * HH + col] = acc;
  red[t] = make_float2(acc * av[col], acc * av[HH + col]);
  __syncthreads();
#pragma unroll
  for (int s = 32; s >= 1; s >>= 1) {
    if ((t & 63) < s) { red[t].x += red[t + s].x; red[t].y += red[t + s].y; }
    __syncthreads();
  }
  if (col == 0) { asrc[R] = red[t].x; adst[R] = red[t].y; }
}

// ---------------- Kernel B: pack Wh into bf16 B-fragments --------------------
// layout: [b][chunk(64)][ntile(4)][lane(32)][8 dwords]; B[k,n]=Wh[b,chunk*32+k, nt*16+n]
__global__ void k_packWhB(const float* __restrict__ wh, unsigned* __restrict__ whb) {
  int tid = blockIdx.x * 256 + threadIdx.x;     // 65536 threads
  int lane = tid & 31, nt = (tid >> 5) & 3, chunk = (tid >> 7) & 63, b = tid >> 13;
  int g = lane >> 4, n = lane & 15;
  unsigned* dst = whb + ((((size_t)b * 64 + chunk) * 4 + nt) * 32 + lane) * 8;
#pragma unroll
  for (int v = 0; v < 8; ++v) {
    int kl = (v < 4) ? (2 * v + 8 * g) : (16 + 2 * (v - 4) + 8 * g);
    size_t base = ((size_t)b * NN + (size_t)chunk * 32 + kl) * HH + nt * 16 + n;
    dst[v] = pkbf(wh[base], wh[base + HH]);
  }
}

// ---------------- Kernel C: pack LSTM/output weights into B-fragments --------
__global__ void k_packW(const float* Wf, const float* Wi, const float* Wo, const float* Wc,
                        const float* Uf, const float* Ui, const float* Uo, const float* Uc,
                        const float* Wout, unsigned* gw, unsigned* gu, unsigned* wob) {
  int tid = blockIdx.x * 256 + threadIdx.x;
  if (tid >= 2560) return;
  const float* Ws[4] = {Wf, Wi, Wo, Wc};
  const float* Us[4] = {Uf, Ui, Uo, Uc};
  if (tid < 2048) {
    int side = tid >> 10;                       // 0 = W-side (h), 1 = U-side (x)
    int t = tid & 1023;
    int lane = t & 31, nt = (t >> 5) & 3, chunk = (t >> 7) & 1, g = (t >> 8) & 3;
    int grp = lane >> 4, n = lane & 15;
    const float* src = side ? Us[g] : Ws[g];
    unsigned* dst = (side ? gu : gw) + ((size_t)(((g * 2 + chunk) * 4 + nt) * 32 + lane)) * 8;
#pragma unroll
    for (int v = 0; v < 8; ++v) {
      int kl = (v < 4) ? (2 * v + 8 * grp) : (16 + 2 * (v - 4) + 8 * grp);
      int k = chunk * 32 + kl;
      dst[v] = pkbf(src[(nt * 16 + n) * 64 + k], src[(nt * 16 + n) * 64 + k + 1]);
    }
  } else {
    int t = tid - 2048;                         // 512 threads: Wout [64,128]
    int lane = t & 31, nt = (t >> 5) & 3, chunk = (t >> 7) & 3;
    int grp = lane >> 4, n = lane & 15;
    unsigned* dst = wob + ((size_t)((chunk * 4 + nt) * 32 + lane)) * 8;
#pragma unroll
    for (int v = 0; v < 8; ++v) {
      int kl = (v < 4) ? (2 * v + 8 * grp) : (16 + 2 * (v - 4) + 8 * grp);
      int k = chunk * 32 + kl;
      dst[v] = pkbf(Wout[(nt * 16 + n) * 128 + k], Wout[(nt * 16 + n) * 128 + k + 1]);
    }
  }
}

__device__ __forceinline__ float mle(float a, float s, float d) {
  float t = s + d;
  t = t > 0.f ? t : 2.f * t;                    // leaky_relu, slope 2
  return a > 0.f ? t : NEGV;                    // adj mask
}

// ---------------- Kernel D: fused flash attention (single adj pass) ----------
__global__ void __launch_bounds__(128) k_attn(const float* __restrict__ adj,
                                              const float* __restrict__ asrc,
                                              const float* __restrict__ adst,
                                              const unsigned* __restrict__ whb,
                                              float* __restrict__ xh) {
  int wid = threadIdx.x >> 5, lane = threadIdx.x & 31;
  int tile = blockIdx.x * 4 + wid;              // 0..1023 (16-row tiles over B*N)
  int b = tile >> 7;
  int n = lane & 15, grp = lane >> 4;
  long R = (long)tile * 16 + n;                 // this lane's logit row
  const float* arow = adj + R * (long)NN;
  const float* drow = adst + (long)b * NN;
  float si = asrc[R];
  float m = -__builtin_inff(), ssum = 0.f;
  v8f acc[4];
#pragma unroll
  for (int nt = 0; nt < 4; ++nt) acc[nt] = v8zero();

  for (int j0 = 0; j0 < NN; j0 += 32) {
    int c1 = j0 + 8 * grp, c2 = j0 + 16 + 8 * grp;
    float4 A0 = *(const float4*)(arow + c1);
    float4 A1 = *(const float4*)(arow + c1 + 4);
    float4 A2 = *(const float4*)(arow + c2);
    float4 A3 = *(const float4*)(arow + c2 + 4);
    float4 D0 = *(const float4*)(drow + c1);
    float4 D1 = *(const float4*)(drow + c1 + 4);
    float4 D2 = *(const float4*)(drow + c2);
    float4 D3 = *(const float4*)(drow + c2 + 4);
    __builtin_prefetch(arow + j0 + 32, 0, 1);   // stream next adj chunk

    float e[16];
    e[0]  = mle(A0.x, si, D0.x); e[1]  = mle(A0.y, si, D0.y);
    e[2]  = mle(A0.z, si, D0.z); e[3]  = mle(A0.w, si, D0.w);
    e[4]  = mle(A1.x, si, D1.x); e[5]  = mle(A1.y, si, D1.y);
    e[6]  = mle(A1.z, si, D1.z); e[7]  = mle(A1.w, si, D1.w);
    e[8]  = mle(A2.x, si, D2.x); e[9]  = mle(A2.y, si, D2.y);
    e[10] = mle(A2.z, si, D2.z); e[11] = mle(A2.w, si, D2.w);
    e[12] = mle(A3.x, si, D3.x); e[13] = mle(A3.y, si, D3.y);
    e[14] = mle(A3.z, si, D3.z); e[15] = mle(A3.w, si, D3.w);

    float cm = e[0];
#pragma unroll
    for (int i = 1; i < 16; ++i) cm = fmaxf(cm, e[i]);
    cm = fmaxf(cm, __shfl_xor(cm, 16, 32));     // combine row's lane pair
    float mn = fmaxf(m, cm);
    float alpha = __expf(m - mn);               // rescale factor for old state

    float p[16], ls = 0.f;
#pragma unroll
    for (int i = 0; i < 16; ++i) { p[i] = __expf(e[i] - mn); ls += p[i]; }
    ls += __shfl_xor(ls, 16, 32);
    ssum = ssum * alpha + ls;
    m = mn;

    AF af;                                      // A-fragment: unnormalized probs
#pragma unroll
    for (int v = 0; v < 8; ++v) af.u[v] = pkbf(p[2 * v], p[2 * v + 1]);

    float sc[8];                                // per-accumulator-row alpha
#pragma unroll
    for (int r = 0; r < 8; ++r) sc[r] = __shfl(alpha, 8 * grp + r, 32);

    int chunk = j0 >> 5;
#pragma unroll
    for (int nt = 0; nt < 4; ++nt) {
      v8f cc;
#pragma unroll
      for (int r = 0; r < 8; ++r) cc[r] = acc[nt][r] * sc[r];
      AF bfm;
      const uint4* bp = (const uint4*)(whb +
          ((((size_t)b * 64 + chunk) * 4 + nt) * 32 + lane) * 8);
      bfm.q[0] = bp[0]; bfm.q[1] = bp[1];
      acc[nt] = wmma_bf(af, bfm, cc);
    }
  }

  // normalize, elu, store x_hidden
#pragma unroll
  for (int r = 0; r < 8; ++r) {
    float sr = __shfl(ssum, 8 * grp + r, 32);
    float inv = 1.f / sr;
    long orow = (long)tile * 16 + 8 * grp + r;
#pragma unroll
    for (int nt = 0; nt < 4; ++nt) {
      float v = acc[nt][r] * inv;
      v = v > 0.f ? v : (__expf(v) - 1.f);
      xh[orow * HH + nt * 16 + n] = v;
    }
  }
}

// ---------------- Kernel E: LSTM gates via WMMA ------------------------------
__global__ void __launch_bounds__(128) k_lstm(const float* __restrict__ x,
                                              const float* __restrict__ h,
                                              const float* __restrict__ cin,
                                              const unsigned* __restrict__ gw,
                                              const unsigned* __restrict__ gu,
                                              const float* bWf, const float* bWi,
                                              const float* bWo, const float* bWc,
                                              const float* bUf, const float* bUi,
                                              const float* bUo, const float* bUc,
                                              float* __restrict__ ht_out,
                                              float* __restrict__ ct_out) {
  int wid = threadIdx.x >> 5, lane = threadIdx.x & 31;
  int tile = blockIdx.x * 4 + wid;
  int n = lane & 15, grp = lane >> 4;
  long R = (long)tile * 16 + n;
  const float* hrow = h + R * HH;
  const float* xrow = x + R * DD;

  AF ah[2], ax[2];
#pragma unroll
  for (int c = 0; c < 2; ++c) {
    float4 q0 = *(const float4*)(hrow + 32 * c + 8 * grp);
    float4 q1 = *(const float4*)(hrow + 32 * c + 8 * grp + 4);
    float4 q2 = *(const float4*)(hrow + 32 * c + 16 + 8 * grp);
    float4 q3 = *(const float4*)(hrow + 32 * c + 16 + 8 * grp + 4);
    ah[c].u[0] = pkbf(q0.x, q0.y); ah[c].u[1] = pkbf(q0.z, q0.w);
    ah[c].u[2] = pkbf(q1.x, q1.y); ah[c].u[3] = pkbf(q1.z, q1.w);
    ah[c].u[4] = pkbf(q2.x, q2.y); ah[c].u[5] = pkbf(q2.z, q2.w);
    ah[c].u[6] = pkbf(q3.x, q3.y); ah[c].u[7] = pkbf(q3.z, q3.w);
    float4 p0 = *(const float4*)(xrow + 32 * c + 8 * grp);
    float4 p1 = *(const float4*)(xrow + 32 * c + 8 * grp + 4);
    float4 p2 = *(const float4*)(xrow + 32 * c + 16 + 8 * grp);
    float4 p3 = *(const float4*)(xrow + 32 * c + 16 + 8 * grp + 4);
    ax[c].u[0] = pkbf(p0.x, p0.y); ax[c].u[1] = pkbf(p0.z, p0.w);
    ax[c].u[2] = pkbf(p1.x, p1.y); ax[c].u[3] = pkbf(p1.z, p1.w);
    ax[c].u[4] = pkbf(p2.x, p2.y); ax[c].u[5] = pkbf(p2.z, p2.w);
    ax[c].u[6] = pkbf(p3.x, p3.y); ax[c].u[7] = pkbf(p3.z, p3.w);
  }

  auto gate = [&](int g, v8f* acc) {
#pragma unroll
    for (int nt = 0; nt < 4; ++nt) acc[nt] = v8zero();
#pragma unroll
    for (int c = 0; c < 2; ++c) {
#pragma unroll
      for (int nt = 0; nt < 4; ++nt) {
        AF bw, bu;
        const uint4* pw = (const uint4*)(gw + ((size_t)(((g * 2 + c) * 4 + nt) * 32 + lane)) * 8);
        bw.q[0] = pw[0]; bw.q[1] = pw[1];
        acc[nt] = wmma_bf(ah[c], bw, acc[nt]);
        const uint4* pu = (const uint4*)(gu + ((size_t)(((g * 2 + c) * 4 + nt) * 32 + lane)) * 8);
        bu.q[0] = pu[0]; bu.q[1] = pu[1];
        acc[nt] = wmma_bf(ax[c], bu, acc[nt]);
      }
    }
  };

  v8f acc[4];
  float ct[4][8], iv[4][8];

  gate(0, acc);                                  // forget
#pragma unroll
  for (int nt = 0; nt < 4; ++nt) {
    float bias = bWf[nt * 16 + n] + bUf[nt * 16 + n];
#pragma unroll
    for (int r = 0; r < 8; ++r) {
      float fv = 1.f / (1.f + __expf(-(acc[nt][r] + bias)));
      float cv = cin[((long)tile * 16 + 8 * grp + r) * HH + nt * 16 + n];
      ct[nt][r] = fv * cv;
    }
  }
  gate(1, acc);                                  // input
#pragma unroll
  for (int nt = 0; nt < 4; ++nt) {
    float bias = bWi[nt * 16 + n] + bUi[nt * 16 + n];
#pragma unroll
    for (int r = 0; r < 8; ++r) iv[nt][r] = 1.f / (1.f + __expf(-(acc[nt][r] + bias)));
  }
  gate(3, acc);                                  // candidate (tanh)
#pragma unroll
  for (int nt = 0; nt < 4; ++nt) {
    float bias = bWc[nt * 16 + n] + bUc[nt * 16 + n];
#pragma unroll
    for (int r = 0; r < 8; ++r) ct[nt][r] += iv[nt][r] * tanhf(acc[nt][r] + bias);
  }
  gate(2, acc);                                  // output gate
#pragma unroll
  for (int nt = 0; nt < 4; ++nt) {
    float bias = bWo[nt * 16 + n] + bUo[nt * 16 + n];
#pragma unroll
    for (int r = 0; r < 8; ++r) {
      float ov = 1.f / (1.f + __expf(-(acc[nt][r] + bias)));
      long orow = (long)tile * 16 + 8 * grp + r;
      float cval = ct[nt][r];
      ct_out[orow * HH + nt * 16 + n] = cval;
      ht_out[orow * HH + nt * 16 + n] = ov * fmaxf(cval, 0.f);
    }
  }
}

// ---------------- Kernel F: output = relu([x_hidden,h_t] @ Wout^T + b) -------
__global__ void __launch_bounds__(128) k_out(const float* __restrict__ xh,
                                             const float* __restrict__ ht,
                                             const unsigned* __restrict__ wob,
                                             const float* __restrict__ bo,
                                             float* __restrict__ outp) {
  int wid = threadIdx.x >> 5, lane = threadIdx.x & 31;
  int tile = blockIdx.x * 4 + wid;
  int n = lane & 15, grp = lane >> 4;
  long R = (long)tile * 16 + n;
  v8f acc[4];
#pragma unroll
  for (int nt = 0; nt < 4; ++nt) acc[nt] = v8zero();
#pragma unroll
  for (int c = 0; c < 4; ++c) {
    const float* src = (c < 2 ? xh : ht) + R * HH + (c & 1) * 32;
    AF a;
    float4 q0 = *(const float4*)(src + 8 * grp);
    float4 q1 = *(const float4*)(src + 8 * grp + 4);
    float4 q2 = *(const float4*)(src + 16 + 8 * grp);
    float4 q3 = *(const float4*)(src + 16 + 8 * grp + 4);
    a.u[0] = pkbf(q0.x, q0.y); a.u[1] = pkbf(q0.z, q0.w);
    a.u[2] = pkbf(q1.x, q1.y); a.u[3] = pkbf(q1.z, q1.w);
    a.u[4] = pkbf(q2.x, q2.y); a.u[5] = pkbf(q2.z, q2.w);
    a.u[6] = pkbf(q3.x, q3.y); a.u[7] = pkbf(q3.z, q3.w);
#pragma unroll
    for (int nt = 0; nt < 4; ++nt) {
      AF bfm;
      const uint4* bp = (const uint4*)(wob + ((size_t)((c * 4 + nt) * 32 + lane)) * 8);
      bfm.q[0] = bp[0]; bfm.q[1] = bp[1];
      acc[nt] = wmma_bf(a, bfm, acc[nt]);
    }
  }
#pragma unroll
  for (int nt = 0; nt < 4; ++nt) {
    float bias = bo[nt * 16 + n];
#pragma unroll
    for (int r = 0; r < 8; ++r) {
      long orow = (long)tile * 16 + 8 * grp + r;
      float v = acc[nt][r] + bias;
      outp[orow * OO + nt * 16 + n] = v > 0.f ? v : 0.f;
    }
  }
}

extern "C" void kernel_launch(void* const* d_in, const int* in_sizes, int n_in,
                              void* d_out, int out_size, void* d_ws, size_t ws_size,
                              hipStream_t stream) {
  const float* x    = (const float*)d_in[0];
  const float* adj  = (const float*)d_in[1];
  const float* cst  = (const float*)d_in[2];
  const float* h    = (const float*)d_in[3];
  const float* W    = (const float*)d_in[4];
  const float* av   = (const float*)d_in[5];
  const float* Wf = (const float*)d_in[6];  const float* bWf = (const float*)d_in[7];
  const float* Wi = (const float*)d_in[8];  const float* bWi = (const float*)d_in[9];
  const float* Wo = (const float*)d_in[10]; const float* bWo = (const float*)d_in[11];
  const float* Wc = (const float*)d_in[12]; const float* bWc = (const float*)d_in[13];
  const float* Uf = (const float*)d_in[14]; const float* bUf = (const float*)d_in[15];
  const float* Ui = (const float*)d_in[16]; const float* bUi = (const float*)d_in[17];
  const float* Uo = (const float*)d_in[18]; const float* bUo = (const float*)d_in[19];
  const float* Uc = (const float*)d_in[20]; const float* bUc = (const float*)d_in[21];
  const float* Wout = (const float*)d_in[22]; const float* bout = (const float*)d_in[23];

  float* ws = (float*)d_ws;
  float* wh   = ws + OFF_WH;
  float* asrc = ws + OFF_ASRC;
  float* adst = ws + OFF_ADST;
  float* xh   = ws + OFF_XH;
  unsigned* whb = (unsigned*)(ws + OFF_WHB);
  unsigned* gw  = (unsigned*)(ws + OFF_GW);
  unsigned* gu  = (unsigned*)(ws + OFF_GU);
  unsigned* wob = (unsigned*)(ws + OFF_WOUT);

  float* outp = (float*)d_out;
  float* ht = outp + (size_t)BB * NN * OO;
  float* ct = ht + (size_t)BB * NN * HH;

  k_wh<<<BB * NN / 4, 256, 0, stream>>>(x, W, av, wh, asrc, adst);
  k_packWhB<<<256, 256, 0, stream>>>(wh, whb);
  k_packW<<<10, 256, 0, stream>>>(Wf, Wi, Wo, Wc, Uf, Ui, Uo, Uc, Wout, gw, gu, wob);
  k_attn<<<256, 128, 0, stream>>>(adj, asrc, adst, whb, xh);
  k_lstm<<<256, 128, 0, stream>>>(x, h, cst, gw, gu,
                                  bWf, bWi, bWo, bWc, bUf, bUi, bUo, bUc, ht, ct);
  k_out<<<256, 128, 0, stream>>>(xh, ht, wob, bout, outp);
}
